// MyNewGCN_25890062860834
// MI455X (gfx1250) — matrix operand order repaired and checked by reading, not compiled, not run on hardware
//
#include <hip/hip_runtime.h>
#include <hip/hip_bf16.h>
#include <math.h>

// Problem constants (match reference)
#define CN  8000     // nodes per side (N1 == N2)
#define CE  128000   // edges per side
#define CB  128      // graphs in batch
#define CD  1024     // Set2Set input dim (2*NCLS)

typedef __attribute__((ext_vector_type(16))) __bf16         v16bf;
typedef __attribute__((ext_vector_type(16))) unsigned short v16u;
typedef __attribute__((ext_vector_type(8)))  unsigned short u16x8;
typedef __attribute__((ext_vector_type(8)))  float          v8f;

union Frag16 { v16u u; v16bf b; u16x8 h[2]; };

__device__ __forceinline__ unsigned short f2bf(float f) {
  unsigned int u = __float_as_uint(f);
  u = (u + 0x7FFFu + ((u >> 16) & 1u)) >> 16;   // round-to-nearest-even
  return (unsigned short)u;
}
__device__ __forceinline__ float sigm(float x) { return 1.0f / (1.0f + __expf(-x)); }

// ---------------------------------------------------------------------------
// bf16 WMMA GEMM, K-contiguous operands only:
//   A[m,k] = A[m*sar + k]        (row-major, K contiguous)
//   B[k,n] = Bt[n*sbc + k]       (column-major storage, K contiguous)
// One wave computes a 32x64 C tile: 2 A-frags x 4 B-frags -> 8 WMMAs per
// K-step, all 6 fragments preloaded so only the first WMMA of the burst
// waits on loadcnt. Per ISA 7.12.2 the per-lane fragment halves are
// contiguous 16B chunks -> global_load_b128.
// ---------------------------------------------------------------------------
__global__ void __launch_bounds__(256)
k_wmma_gemm(const unsigned short* __restrict__ A, long sar,
            const unsigned short* __restrict__ Bm, long sbc,
            float* __restrict__ C, long ldc,
            const float* __restrict__ bias, int relu,
            int tilesM32, int tilesN64, int K)
{
  int tile = blockIdx.x * 8 + threadIdx.y;
  if (tile >= tilesM32 * tilesN64) return;
  int tm = tile / tilesN64, tn = tile % tilesN64;
  int lane = threadIdx.x;
  int l15 = lane & 15, half = lane >> 4;

  long arow0 = (long)(tm * 32 + l15) * sar;
  long arow1 = arow0 + 16 * sar;
  long bcol[4];
#pragma unroll
  for (int j = 0; j < 4; ++j) bcol[j] = (long)(tn * 64 + j * 16 + l15) * sbc;

  v8f acc[2][4] = {{{}, {}, {}, {}}, {{}, {}, {}, {}}};
  for (int k0 = 0; k0 < K; k0 += 32) {
    Frag16 a0, a1, b[4];
    const u16x8* ap0 = (const u16x8*)(A + arow0 + k0 + 8 * half);
    const u16x8* ap1 = (const u16x8*)(A + arow1 + k0 + 8 * half);
    a0.h[0] = ap0[0]; a0.h[1] = ap0[2];   // K = k0+8h..+7, k0+16+8h..+7
    a1.h[0] = ap1[0]; a1.h[1] = ap1[2];
#pragma unroll
    for (int j = 0; j < 4; ++j) {
      const u16x8* bp = (const u16x8*)(Bm + bcol[j] + k0 + 8 * half);
      b[j].h[0] = bp[0];                  // K = k0+8h..+7
      b[j].h[1] = bp[2];                  // K = k0+16+8h..+7
    }
    __builtin_prefetch(A + arow0 + k0 + 128, 0, 1);    // global_prefetch_b8
    __builtin_prefetch(A + arow1 + k0 + 128, 0, 1);
#pragma unroll
    for (int j = 0; j < 4; ++j)
      __builtin_prefetch(Bm + bcol[j] + k0 + 128, 0, 1);
#pragma unroll
    for (int j = 0; j < 4; ++j) {
      acc[0][j] = __builtin_amdgcn_wmma_f32_16x16x32_bf16(
          false, a0.b, false, b[j].b, (short)0, acc[0][j], false, false);
      acc[1][j] = __builtin_amdgcn_wmma_f32_16x16x32_bf16(
          false, a1.b, false, b[j].b, (short)0, acc[1][j], false, false);
    }
  }

#pragma unroll
  for (int i = 0; i < 2; ++i) {
    int mbase = tm * 32 + i * 16 + 8 * half;
#pragma unroll
    for (int j = 0; j < 4; ++j) {
      int col = tn * 64 + j * 16 + l15;
      float bv = bias ? bias[col] : 0.0f;
#pragma unroll
      for (int r = 0; r < 8; ++r) {
        float v = acc[i][j][r] + bv;
        if (relu) v = fmaxf(v, 0.0f);
        C[(long)(mbase + r) * ldc + col] = v;
      }
    }
  }
}

// ---------------------------------------------------------------------------
// Interaction map: imap[i,j] = tanh( s[i,:] . v[j,:] ) * (gid1[i]==gid2[j])
// 32x64 tile per wave; writes imap AND imapT (bf16) from registers so both
// consumer GEMMs read K-contiguous data. 256MB bf16 sits mostly in 192MB L2.
// ---------------------------------------------------------------------------
__global__ void __launch_bounds__(256)
k_wmma_imap(const unsigned short* __restrict__ Sb, const unsigned short* __restrict__ Vb,
            const int* __restrict__ g1, const int* __restrict__ g2,
            unsigned short* __restrict__ imap, unsigned short* __restrict__ imapT,
            int tilesM32, int tilesN64, int K)
{
  int tile = blockIdx.x * 8 + threadIdx.y;
  if (tile >= tilesM32 * tilesN64) return;
  int tm = tile / tilesN64, tn = tile % tilesN64;
  int lane = threadIdx.x;
  int l15 = lane & 15, half = lane >> 4;

  long arow0 = (long)(tm * 32 + l15) * K;
  long arow1 = arow0 + (long)16 * K;
  long brow[4];
#pragma unroll
  for (int j = 0; j < 4; ++j) brow[j] = (long)(tn * 64 + j * 16 + l15) * K;

  v8f acc[2][4] = {{{}, {}, {}, {}}, {{}, {}, {}, {}}};
  for (int k0 = 0; k0 < K; k0 += 32) {
    Frag16 a0, a1, b[4];
    const u16x8* ap0 = (const u16x8*)(Sb + arow0 + k0 + 8 * half);
    const u16x8* ap1 = (const u16x8*)(Sb + arow1 + k0 + 8 * half);
    a0.h[0] = ap0[0]; a0.h[1] = ap0[2];
    a1.h[0] = ap1[0]; a1.h[1] = ap1[2];
#pragma unroll
    for (int j = 0; j < 4; ++j) {
      const u16x8* bp = (const u16x8*)(Vb + brow[j] + k0 + 8 * half);
      b[j].h[0] = bp[0];
      b[j].h[1] = bp[2];
    }
    __builtin_prefetch(Sb + arow0 + k0 + 128, 0, 1);
    __builtin_prefetch(Sb + arow1 + k0 + 128, 0, 1);
#pragma unroll
    for (int j = 0; j < 4; ++j) {
      acc[0][j] = __builtin_amdgcn_wmma_f32_16x16x32_bf16(
          false, a0.b, false, b[j].b, (short)0, acc[0][j], false, false);
      acc[1][j] = __builtin_amdgcn_wmma_f32_16x16x32_bf16(
          false, a1.b, false, b[j].b, (short)0, acc[1][j], false, false);
    }
  }

#pragma unroll
  for (int i = 0; i < 2; ++i) {
    int mbase = tm * 32 + i * 16 + 8 * half;
#pragma unroll
    for (int j = 0; j < 4; ++j) {
      int col = tn * 64 + j * 16 + l15;
      int gc = g2[col];
#pragma unroll
      for (int r = 0; r < 8; ++r) {
        int m = mbase + r;
        float v = tanhf(acc[i][j][r]);
        if (g1[m] != gc) v = 0.0f;
        unsigned short bv = f2bf(v);
        imap[(long)m * CN + col] = bv;
        imapT[(long)col * CN + m] = bv;
      }
    }
  }
}

// ------------------------- elementwise / irregular kernels ------------------
__global__ void k_zero_f32(float* p, long n) {
  long i = (long)blockIdx.x * blockDim.x + threadIdx.x;
  if (i < n) p[i] = 0.0f;
}
__global__ void k_zero_u32(unsigned int* p, long n) {
  long i = (long)blockIdx.x * blockDim.x + threadIdx.x;
  if (i < n) p[i] = 0u;
}
__global__ void k_count_deg(const int* __restrict__ src, const int* __restrict__ dst,
                            float* outd, float* ind, int E) {
  int e = blockIdx.x * blockDim.x + threadIdx.x;
  if (e < E) { atomicAdd(&outd[src[e]], 1.0f); atomicAdd(&ind[dst[e]], 1.0f); }
}
__global__ void k_rsqrt_clip(float* d, int n) {
  int i = blockIdx.x * blockDim.x + threadIdx.x;
  if (i < n) d[i] = rsqrtf(fmaxf(d[i], 1.0f));
}
// y[i] = bf16( x[i] * rowscale[i/C] )   (rowscale nullable)
__global__ void k_cvt_bf16(const float* __restrict__ x, const float* __restrict__ rowscale,
                           unsigned short* __restrict__ y, long n, int C) {
  long i = (long)blockIdx.x * blockDim.x + threadIdx.x;
  if (i < n) {
    float s = rowscale ? rowscale[i / C] : 1.0f;
    y[i] = f2bf(x[i] * s);
  }
}
// Wt[n*K + k] = bf16( W[k*N + n] )  : row-major [K,N] -> K-contiguous [N,K]
__global__ void k_cvt_bf16_t(const float* __restrict__ W, unsigned short* __restrict__ Wt,
                             int K, int N) {
  long i = (long)blockIdx.x * blockDim.x + threadIdx.x;
  if (i < (long)K * N) {
    int k = (int)(i / N), n = (int)(i % N);
    Wt[(long)n * K + k] = f2bf(W[i]);
  }
}
// dst[c*R + r] = src[r*C + c]
__global__ void k_transpose_u16(const unsigned short* __restrict__ src,
                                unsigned short* __restrict__ dst, int R, int C) {
  long i = (long)blockIdx.x * blockDim.x + threadIdx.x;
  if (i < (long)R * C) {
    int r = (int)(i / C), c = (int)(i % C);
    dst[(long)c * R + r] = src[i];
  }
}
__global__ void k_scatter_edges(const float* __restrict__ h, const int* __restrict__ src,
                                const int* __restrict__ dst, float* __restrict__ agg,
                                int E, int C) {
  long i = (long)blockIdx.x * blockDim.x + threadIdx.x;
  if (i < (long)E * C) {
    int e = (int)(i / C), c = (int)(i % C);
    atomicAdd(&agg[(long)dst[e] * C + c], h[(long)src[e] * C + c]);
  }
}
__global__ void k_epilogue_norm(const float* __restrict__ agg, const float* __restrict__ inds,
                                const float* __restrict__ bias, float* __restrict__ out,
                                int N, int C, int relu) {
  long i = (long)blockIdx.x * blockDim.x + threadIdx.x;
  if (i < (long)N * C) {
    int n = (int)(i / C), c = (int)(i % C);
    float v = agg[i] * inds[n] + bias[c];
    if (relu) v = fmaxf(v, 0.0f);
    out[i] = v;
  }
}
__global__ void k_copy_strided(const float* __restrict__ src, long lds,
                               float* __restrict__ dst, long ldd, int rows, int cols) {
  long i = (long)blockIdx.x * blockDim.x + threadIdx.x;
  if (i < (long)rows * cols) {
    long r = i / cols, c = i % cols;
    dst[r * ldd + c] = src[r * lds + c];
  }
}
// LSTM cell, torch gate order i,f,g,o over 4*H, H = CD
__global__ void k_lstm_cell(const float* __restrict__ gq, const float* __restrict__ gh,
                            const float* __restrict__ bih, const float* __restrict__ bhh,
                            float* __restrict__ c, float* __restrict__ h) {
  int i = blockIdx.x * blockDim.x + threadIdx.x;
  if (i < CB * CD) {
    int b = i / CD, u = i % CD;
    long base = (long)b * 4 * CD;
    float gi = gq[base + u]        + gh[base + u]        + bih[u]        + bhh[u];
    float gf = gq[base + CD + u]   + gh[base + CD + u]   + bih[CD + u]   + bhh[CD + u];
    float gg = gq[base + 2*CD + u] + gh[base + 2*CD + u] + bih[2*CD + u] + bhh[2*CD + u];
    float go = gq[base + 3*CD + u] + gh[base + 3*CD + u] + bih[3*CD + u] + bhh[3*CD + u];
    float cc = sigm(gf) * c[i] + sigm(gi) * tanhf(gg);
    c[i] = cc;
    h[i] = sigm(go) * tanhf(cc);
  }
}
__global__ void k_attn_e(const float* __restrict__ x, const float* __restrict__ h,
                         const int* __restrict__ gid, float* __restrict__ e, int N) {
  int n = blockIdx.x * blockDim.x + threadIdx.x;
  if (n < N) {
    const float* xr = x + (long)n * CD;
    const float* hr = h + (long)gid[n] * CD;
    float acc = 0.0f;
    for (int k = 0; k < CD; ++k) acc += xr[k] * hr[k];
    e[n] = acc;
  }
}
__device__ __forceinline__ unsigned int f_ord(float f) {
  unsigned int u = __float_as_uint(f);
  return (u & 0x80000000u) ? ~u : (u | 0x80000000u);
}
__device__ __forceinline__ float ord_f(unsigned int u) {
  return (u & 0x80000000u) ? __uint_as_float(u & 0x7FFFFFFFu) : __uint_as_float(~u);
}
__global__ void k_seg_max(const float* __restrict__ e, const int* __restrict__ gid,
                          unsigned int* __restrict__ mu, int N) {
  int n = blockIdx.x * blockDim.x + threadIdx.x;
  if (n < N) atomicMax(&mu[gid[n]], f_ord(e[n]));
}
__global__ void k_seg_exp_sum(const float* __restrict__ e, const unsigned int* __restrict__ mu,
                              const int* __restrict__ gid, float* __restrict__ ex,
                              float* __restrict__ ssum, int N) {
  int n = blockIdx.x * blockDim.x + threadIdx.x;
  if (n < N) {
    float v = __expf(e[n] - ord_f(mu[gid[n]]));
    ex[n] = v;
    atomicAdd(&ssum[gid[n]], v);
  }
}
__global__ void k_seg_wsum(const float* __restrict__ x, const float* __restrict__ ex,
                           const float* __restrict__ ssum, const int* __restrict__ gid,
                           float* __restrict__ r, int N) {
  long i = (long)blockIdx.x * blockDim.x + threadIdx.x;
  if (i < (long)N * CD) {
    int n = (int)(i / CD), k = (int)(i % CD);
    int g = gid[n];
    atomicAdd(&r[(long)g * CD + k], (ex[n] / ssum[g]) * x[i]);
  }
}
__global__ void k_qstar(const float* __restrict__ h, const float* __restrict__ r,
                        float* __restrict__ q) {
  int i = blockIdx.x * blockDim.x + threadIdx.x;
  if (i < CB * 2 * CD) {
    int b = i / (2 * CD), j = i % (2 * CD);
    q[i] = (j < CD) ? h[(long)b * CD + j] : r[(long)b * CD + (j - CD)];
  }
}
__global__ void k_fc3(const float* __restrict__ z2, const float* __restrict__ w,
                      const float* __restrict__ b, float* __restrict__ out) {
  int i = threadIdx.x;  // 128
  float acc = 0.0f;
  for (int k = 0; k < 512; ++k) acc += z2[(long)i * 512 + k] * w[k];
  out[i] = acc + b[0];
}

// ---------------------------------------------------------------------------
extern "C" void kernel_launch(void* const* d_in, const int* in_sizes, int n_in,
                              void* d_out, int out_size, void* d_ws, size_t ws_size,
                              hipStream_t stream) {
  (void)in_sizes; (void)n_in; (void)out_size; (void)ws_size;
  const float* x1   = (const float*)d_in[0];
  const float* x2   = (const float*)d_in[1];
  const int*   src1 = (const int*)d_in[2];
  const int*   dst1 = (const int*)d_in[3];
  const int*   src2 = (const int*)d_in[4];
  const int*   dst2 = (const int*)d_in[5];
  const int*   gid1 = (const int*)d_in[6];
  const int*   gid2 = (const int*)d_in[7];
  const float* Wgc1 = (const float*)d_in[8];  const float* bgc1 = (const float*)d_in[9];
  const float* Wgc2 = (const float*)d_in[10]; const float* bgc2 = (const float*)d_in[11];
  const float* Wgc3 = (const float*)d_in[12]; const float* bgc3 = (const float*)d_in[13];
  const float* Wgc4 = (const float*)d_in[14]; const float* bgc4 = (const float*)d_in[15];
  const float* Wih  = (const float*)d_in[16]; const float* Whh  = (const float*)d_in[17];
  const float* bih  = (const float*)d_in[18]; const float* bhh  = (const float*)d_in[19];
  const float* Wfc1 = (const float*)d_in[20]; const float* bfc1 = (const float*)d_in[21];
  const float* Wfc2 = (const float*)d_in[22]; const float* bfc2 = (const float*)d_in[23];
  const float* Wfc3 = (const float*)d_in[24]; const float* bfc3 = (const float*)d_in[25];

  // ---- workspace linear allocator ----
  char* wsp = (char*)d_ws; size_t off = 0;
  auto alloc = [&](size_t bytes) -> void* {
    void* p = wsp + off; off = (off + bytes + 255) & ~(size_t)255; return p;
  };
  float* deg1o = (float*)alloc(CN * 4); float* deg1i = (float*)alloc(CN * 4);
  float* deg2o = (float*)alloc(CN * 4); float* deg2i = (float*)alloc(CN * 4);
  unsigned short* bfin = (unsigned short*)alloc((size_t)CN * 512 * 2);
  float* hbuf = (float*)alloc((size_t)CN * 512 * 4);
  float* aggb = (float*)alloc((size_t)CN * 512 * 4);
  float* t1   = (float*)alloc((size_t)CN * 512 * 4);
  float* sM   = (float*)alloc((size_t)CN * 512 * 4);
  float* vM   = (float*)alloc((size_t)CN * 512 * 4);
  unsigned short* sb  = (unsigned short*)alloc((size_t)CN * 512 * 2);
  unsigned short* vb  = (unsigned short*)alloc((size_t)CN * 512 * 2);
  unsigned short* sbT = (unsigned short*)alloc((size_t)CN * 512 * 2);
  unsigned short* vbT = (unsigned short*)alloc((size_t)CN * 512 * 2);
  unsigned short* imap  = (unsigned short*)alloc((size_t)CN * CN * 2);   // 128 MB
  unsigned short* imapT = (unsigned short*)alloc((size_t)CN * CN * 2);   // 128 MB
  float* sf = (float*)alloc((size_t)CN * 1024 * 4);
  float* vf = (float*)alloc((size_t)CN * 1024 * 4);
  unsigned short* wgc1t = (unsigned short*)alloc(128 * 256 * 2);
  unsigned short* wgc2t = (unsigned short*)alloc(256 * 512 * 2);
  unsigned short* wgc3t = (unsigned short*)alloc(128 * 256 * 2);
  unsigned short* wgc4t = (unsigned short*)alloc(256 * 512 * 2);
  unsigned short* wihb  = (unsigned short*)alloc((size_t)4096 * 2048 * 2);
  unsigned short* whhb  = (unsigned short*)alloc((size_t)4096 * 1024 * 2);
  unsigned short* wfc1b = (unsigned short*)alloc((size_t)1024 * 4096 * 2);
  unsigned short* wfc2b = (unsigned short*)alloc((size_t)512 * 1024 * 2);
  float* qstar = (float*)alloc((size_t)CB * 2 * CD * 4);
  float* hl    = (float*)alloc((size_t)CB * CD * 4);
  float* cl    = (float*)alloc((size_t)CB * CD * 4);
  float* gq    = (float*)alloc((size_t)CB * 4 * CD * 4);
  float* gh    = (float*)alloc((size_t)CB * 4 * CD * 4);
  float* ebuf  = (float*)alloc(CN * 4);
  float* exb   = (float*)alloc(CN * 4);
  unsigned int* mu = (unsigned int*)alloc(CB * 4);
  float* ssum  = (float*)alloc(CB * 4);
  float* rbuf  = (float*)alloc((size_t)CB * CD * 4);
  unsigned short* qsb = (unsigned short*)alloc((size_t)CB * 2 * CD * 2);
  unsigned short* hlb = (unsigned short*)alloc((size_t)CB * CD * 2);
  float* z  = (float*)alloc((size_t)CB * 4096 * 4);
  unsigned short* zb  = (unsigned short*)alloc((size_t)CB * 4096 * 2);
  float* z1 = (float*)alloc((size_t)CB * 1024 * 4);
  unsigned short* z1b = (unsigned short*)alloc((size_t)CB * 1024 * 2);
  float* z2 = (float*)alloc((size_t)CB * 512 * 4);

  // ---- helpers ----
  auto zf = [&](float* p, long n) {
    k_zero_f32<<<dim3((unsigned)((n + 255) / 256)), 256, 0, stream>>>(p, n);
  };
  auto cvt = [&](const float* x, const float* rs, unsigned short* y, long n, int C) {
    k_cvt_bf16<<<dim3((unsigned)((n + 255) / 256)), 256, 0, stream>>>(x, rs, y, n, C);
  };
  auto cvt_t = [&](const float* W, unsigned short* Wt, int K, int N) {
    long n = (long)K * N;
    k_cvt_bf16_t<<<dim3((unsigned)((n + 255) / 256)), 256, 0, stream>>>(W, Wt, K, N);
  };
  // C[m,n] = sum_k A[m*sar+k] * Bt[n*sbc+k]; tiles: 32 rows x 64 cols per wave
  auto gemm = [&](const unsigned short* A, long sar,
                  const unsigned short* Bt, long sbc,
                  float* C, long ldc, const float* bias, int relu,
                  int M, int N, int K) {
    int tiles = (M / 32) * (N / 64);
    k_wmma_gemm<<<dim3((unsigned)((tiles + 7) / 8)), dim3(32, 8), 0, stream>>>(
        A, sar, Bt, sbc, C, ldc, bias, relu, M / 32, N / 64, K);
  };

  // ---- degrees ----
  zf(deg1o, CN); zf(deg1i, CN); zf(deg2o, CN); zf(deg2i, CN);
  k_count_deg<<<(CE + 255) / 256, 256, 0, stream>>>(src1, dst1, deg1o, deg1i, CE);
  k_count_deg<<<(CE + 255) / 256, 256, 0, stream>>>(src2, dst2, deg2o, deg2i, CE);
  k_rsqrt_clip<<<(CN + 255) / 256, 256, 0, stream>>>(deg1o, CN);
  k_rsqrt_clip<<<(CN + 255) / 256, 256, 0, stream>>>(deg1i, CN);
  k_rsqrt_clip<<<(CN + 255) / 256, 256, 0, stream>>>(deg2o, CN);
  k_rsqrt_clip<<<(CN + 255) / 256, 256, 0, stream>>>(deg2i, CN);

  // ---- weight conversions to K-contiguous bf16 ----
  cvt_t(Wgc1, wgc1t, 128, 256);              // [N=256][K=128]
  cvt_t(Wgc2, wgc2t, 256, 512);
  cvt_t(Wgc3, wgc3t, 128, 256);
  cvt_t(Wgc4, wgc4t, 256, 512);
  cvt(Wih, nullptr, wihb, (long)4096 * 2048, 1);   // W.T GEMMs: already K-contig
  cvt(Whh, nullptr, whhb, (long)4096 * 1024, 1);
  cvt(Wfc1, nullptr, wfc1b, (long)1024 * 4096, 1);
  cvt(Wfc2, nullptr, wfc2b, (long)512 * 1024, 1);

  // ---- GraphConv: out = segsum((x*outd^-.5 @ W)[src] -> dst) * ind^-.5 + b ----
  auto gconv = [&](const float* x, int Cin, const unsigned short* Wt, int Cout,
                   const float* bias, const int* src, const int* dst,
                   const float* dO, const float* dI, float* out, int relu) {
    cvt(x, dO, bfin, (long)CN * Cin, Cin);
    gemm(bfin, Cin, Wt, Cin, hbuf, Cout, nullptr, 0, CN, Cout, Cin);
    zf(aggb, (long)CN * Cout);
    long ne = (long)CE * Cout;
    k_scatter_edges<<<dim3((unsigned)((ne + 255) / 256)), 256, 0, stream>>>(
        hbuf, src, dst, aggb, CE, Cout);
    long nn = (long)CN * Cout;
    k_epilogue_norm<<<dim3((unsigned)((nn + 255) / 256)), 256, 0, stream>>>(
        aggb, dI, bias, out, CN, Cout, relu);
  };

  gconv(x1, 128, wgc1t, 256, bgc1, src1, dst1, deg1o, deg1i, t1, 1);
  gconv(t1, 256, wgc2t, 512, bgc2, src1, dst1, deg1o, deg1i, sM, 0);
  gconv(x2, 128, wgc3t, 256, bgc3, src2, dst2, deg2o, deg2i, t1, 1);
  gconv(t1, 256, wgc4t, 512, bgc4, src2, dst2, deg2o, deg2i, vM, 0);

  // ---- interaction map + cross features ----
  cvt(sM, nullptr, sb, (long)CN * 512, 1);
  cvt(vM, nullptr, vb, (long)CN * 512, 1);
  {
    long n = (long)CN * 512;
    k_transpose_u16<<<dim3((unsigned)((n + 255) / 256)), 256, 0, stream>>>(sb, sbT, CN, 512);
    k_transpose_u16<<<dim3((unsigned)((n + 255) / 256)), 256, 0, stream>>>(vb, vbT, CN, 512);
  }
  {
    int tiles = (CN / 32) * (CN / 64);
    k_wmma_imap<<<dim3((unsigned)((tiles + 7) / 8)), dim3(32, 8), 0, stream>>>(
        sb, vb, gid1, gid2, imap, imapT, CN / 32, CN / 64, 512);
  }
  {
    long n = (long)CN * 512;
    k_copy_strided<<<dim3((unsigned)((n + 255) / 256)), 256, 0, stream>>>(sM, 512, sf, 1024, CN, 512);
    k_copy_strided<<<dim3((unsigned)((n + 255) / 256)), 256, 0, stream>>>(vM, 512, vf, 1024, CN, 512);
  }
  // solute_prime = imap @ v    -> sf[:, 512:1024]   (B[k,n]=vb[k,n] -> vbT)
  gemm(imap, CN, vbT, CN, sf + 512, 1024, nullptr, 0, CN, 512, CN);
  // solvent_prime = imap.T @ s -> vf[:, 512:1024]   (A=imapT, B[k,n]=sb[k,n] -> sbT)
  gemm(imapT, CN, sbT, CN, vf + 512, 1024, nullptr, 0, CN, 512, CN);

  // ---- Set2Set (n_iters = 2) ----
  auto set2set = [&](const float* xf, const int* gid, float* zdst) {
    zf(qstar, (long)CB * 2 * CD); zf(hl, (long)CB * CD); zf(cl, (long)CB * CD);
    for (int it = 0; it < 2; ++it) {
      cvt(qstar, nullptr, qsb, (long)CB * 2 * CD, 1);
      cvt(hl, nullptr, hlb, (long)CB * CD, 1);
      // gates = q_star @ W_ih.T  and  h @ W_hh.T  (B[k,n] = W[n,k] : K-contig)
      gemm(qsb, 2 * CD, wihb, 2 * CD, gq, 4 * CD, nullptr, 0, CB, 4 * CD, 2 * CD);
      gemm(hlb, CD, whhb, CD, gh, 4 * CD, nullptr, 0, CB, 4 * CD, CD);
      k_lstm_cell<<<(CB * CD + 255) / 256, 256, 0, stream>>>(gq, gh, bih, bhh, cl, hl);
      k_attn_e<<<(CN + 255) / 256, 256, 0, stream>>>(xf, hl, gid, ebuf, CN);
      k_zero_u32<<<1, CB, 0, stream>>>(mu, CB);
      zf(ssum, CB);
      k_seg_max<<<(CN + 255) / 256, 256, 0, stream>>>(ebuf, gid, mu, CN);
      k_seg_exp_sum<<<(CN + 255) / 256, 256, 0, stream>>>(ebuf, mu, gid, exb, ssum, CN);
      zf(rbuf, (long)CB * CD);
      long nw = (long)CN * CD;
      k_seg_wsum<<<dim3((unsigned)((nw + 255) / 256)), 256, 0, stream>>>(xf, exb, ssum, gid, rbuf, CN);
      k_qstar<<<(CB * 2 * CD + 255) / 256, 256, 0, stream>>>(hl, rbuf, qstar);
    }
    long n = (long)CB * 2 * CD;
    k_copy_strided<<<dim3((unsigned)((n + 255) / 256)), 256, 0, stream>>>(qstar, 2 * CD, zdst, 4096, CB, 2 * CD);
  };
  set2set(sf, gid1, z);          // cols [0, 2048)
  set2set(vf, gid2, z + 2048);   // cols [2048, 4096)

  // ---- MLP head ----
  cvt(z, nullptr, zb, (long)CB * 4096, 1);
  gemm(zb, 4096, wfc1b, 4096, z1, 1024, bfc1, 1, CB, 1024, 4096);
  cvt(z1, nullptr, z1b, (long)CB * 1024, 1);
  gemm(z1b, 1024, wfc2b, 1024, z2, 512, bfc2, 1, CB, 512, 1024);
  k_fc3<<<1, CB, 0, stream>>>(z2, Wfc3, bfc3, (float*)d_out);
}
